// EmilyGIN_angle_87703232184760
// MI455X (gfx1250) — matrix operand order repaired and checked by reading, not compiled, not run on hardware
//
#include <hip/hip_runtime.h>
#include <hip/hip_bf16.h>
#include <stdint.h>

#define DD 128
#define MT 4                    // 16-row M-tiles per block -> 64 rows/block
#define ROWS_PER_BLOCK (16 * MT)

typedef __attribute__((ext_vector_type(16))) _Float16 v16h;
typedef __attribute__((ext_vector_type(8)))  float    v8f;

union Frag16 { uint32_t u[8]; v16h h; };

// A-fragment (16x32 f16, ISA 7.12.2): lane L -> m = L%16, g = L/16.
// Pair p holds K = {2p,2p+1}, plus 16*g (matches B-fragment order).
// k-contiguous across all 8 pairs -> compiler merges into ds_load_b128.
__device__ __forceinline__ v16h ldfragA(const _Float16* base, int m, int g, int koff) {
    Frag16 f;
#pragma unroll
    for (int p = 0; p < 8; ++p) {
        int k = 2 * p + 16 * g + koff;
        f.u[p] = *(const uint32_t*)(base + m * DD + k);
    }
    return f.h;
}

// B-fragment (32x16 f16): lane L -> n = L%16, g = L/16. Pair p: K = 2p + 16g.
// B stored transposed ([n][k]) so all 8 pairs are one contiguous 32B run.
__device__ __forceinline__ v16h ldfragB(const _Float16* base, int n, int g, int koff) {
    Frag16 f;
#pragma unroll
    for (int p = 0; p < 8; ++p) {
        int k = 2 * p + 16 * g + koff;
        f.u[p] = *(const uint32_t*)(base + n * DD + k);
    }
    return f.h;
}

// ---------------- kernel 1: agg = feature (h = (1+eps)*x + sum, eps = 0) ----
__global__ __launch_bounds__(256) void gin_init_kernel(const float* __restrict__ feat,
                                                       float* __restrict__ agg, long n4) {
    long i = (long)blockIdx.x * blockDim.x + threadIdx.x;
    if (i < n4) ((float4*)agg)[i] = ((const float4*)feat)[i];
}

// ---------------- kernel 1b: pre-convert + transpose weights to f16 --------
__global__ __launch_bounds__(256) void gin_cvtw_kernel(const float* __restrict__ W1,
                                                       const float* __restrict__ W2,
                                                       _Float16* __restrict__ W1t,
                                                       _Float16* __restrict__ W2t) {
    int i = blockIdx.x * 256 + threadIdx.x;
    if (i < DD * DD) {
        int k = i >> 7, n = i & (DD - 1);
        W1t[n * DD + k] = (_Float16)W1[i];
        W2t[n * DD + k] = (_Float16)W2[i];
    }
}

// ---------------- kernel 2: scatter-add neighbor rows ----------------------
__global__ __launch_bounds__(256) void gin_scatter_kernel(const float* __restrict__ feat,
                                                          const int* __restrict__ src,
                                                          const int* __restrict__ dst,
                                                          float* __restrict__ agg, int nE) {
    int e = blockIdx.x * 8 + (threadIdx.x >> 5);
    if (e >= nE) return;
    int lane = threadIdx.x & 31;
    long s = (long)src[e] * DD + lane * 4;
    long d = (long)dst[e] * DD + lane * 4;
    float4 v = *(const float4*)(feat + s);
    atomicAdd(agg + d + 0, v.x);
    atomicAdd(agg + d + 1, v.y);
    atomicAdd(agg + d + 2, v.z);
    atomicAdd(agg + d + 3, v.w);
}

// ---------------- kernel 3: fused 2-layer MLP via WMMA + ReLU + stat partials
__global__ __launch_bounds__(256) void gin_mlp_wmma_kernel(const float* __restrict__ h,
                                                           const _Float16* __restrict__ W1t,
                                                           const float* __restrict__ b1,
                                                           const _Float16* __restrict__ W2t,
                                                           const float* __restrict__ b2,
                                                           float* __restrict__ y,
                                                           float* __restrict__ pSum,
                                                           float* __restrict__ pSq,
                                                           int nNodes) {
    __shared__ _Float16 sW1t[DD * DD];                 // 32 KB, [n][k]
    __shared__ _Float16 sW2t[DD * DD];                 // 32 KB, [n][k]
    __shared__ _Float16 sA [ROWS_PER_BLOCK * DD];      // 16 KB
    __shared__ _Float16 sH1[ROWS_PER_BLOCK * DD];      // 16 KB
    __shared__ float    sSum[DD];
    __shared__ float    sSq[DD];

    int tid = threadIdx.x;

    // Stage pre-converted weights: straight uint4 copies, no convert, no transpose
    {
        const uint4* w1 = (const uint4*)W1t;
        const uint4* w2 = (const uint4*)W2t;
        uint4* d1 = (uint4*)sW1t;
        uint4* d2 = (uint4*)sW2t;
        const int nvec = DD * DD / 8;                  // 2048 uint4 per matrix
#pragma unroll 4
        for (int i = tid; i < nvec; i += 256) { d1[i] = w1[i]; d2[i] = w2[i]; }
    }
    // Stage A tile, convert f32 -> f16
    long row0 = (long)blockIdx.x * ROWS_PER_BLOCK;
    for (int i = tid; i < ROWS_PER_BLOCK * DD / 4; i += 256) {
        int m  = i >> 5;                               // 32 float4 per row
        int k4 = (i & 31) * 4;
        long r = row0 + m;
        if (r >= nNodes) r = nNodes - 1;               // clamp tail block
        float4 v = *(const float4*)(h + r * DD + k4);
        _Float16* p = sA + m * DD + k4;
        p[0] = (_Float16)v.x; p[1] = (_Float16)v.y;
        p[2] = (_Float16)v.z; p[3] = (_Float16)v.w;
    }
    if (tid < DD) { sSum[tid] = 0.f; sSq[tid] = 0.f; }
    __syncthreads();

    int lane = tid & 31;
    int w    = tid >> 5;       // 8 waves; wave w owns output columns [16w, 16w+16)
    int g    = lane >> 4;
    int lm   = lane & 15;
    int nw   = w * 16;
    int n    = nw + lm;        // this lane's output column

    // ---- GEMM 1: (64xDD) @ W1, B-fragment reused across 4 M-tiles ----
    v8f c1[MT];
#pragma unroll
    for (int mt = 0; mt < MT; ++mt)
        c1[mt] = (v8f){0.f, 0.f, 0.f, 0.f, 0.f, 0.f, 0.f, 0.f};
#pragma unroll
    for (int kt = 0; kt < 4; ++kt) {
        v16h b = ldfragB(sW1t + nw * DD, lm, g, kt * 32);
#pragma unroll
        for (int mt = 0; mt < MT; ++mt) {
            v16h a = ldfragA(sA + mt * 16 * DD, lm, g, kt * 32);
            c1[mt] = __builtin_amdgcn_wmma_f32_16x16x32_f16(false, a, false, b,
                                                            (short)0, c1[mt], false, false);
        }
    }
    float bias1 = b1[n];
#pragma unroll
    for (int mt = 0; mt < MT; ++mt) {
#pragma unroll
        for (int r = 0; r < 8; ++r) {
            int m = mt * 16 + r + 8 * g;
            float v = c1[mt][r] + bias1;
            v = v > 0.f ? v : 0.f;                     // inner ReLU
            sH1[m * DD + n] = (_Float16)v;
        }
    }
    __syncthreads();

    // ---- GEMM 2: (64xDD) @ W2 ----
    v8f c2[MT];
#pragma unroll
    for (int mt = 0; mt < MT; ++mt)
        c2[mt] = (v8f){0.f, 0.f, 0.f, 0.f, 0.f, 0.f, 0.f, 0.f};
#pragma unroll
    for (int kt = 0; kt < 4; ++kt) {
        v16h b = ldfragB(sW2t + nw * DD, lm, g, kt * 32);
#pragma unroll
        for (int mt = 0; mt < MT; ++mt) {
            v16h a = ldfragA(sH1 + mt * 16 * DD, lm, g, kt * 32);
            c2[mt] = __builtin_amdgcn_wmma_f32_16x16x32_f16(false, a, false, b,
                                                            (short)0, c2[mt], false, false);
        }
    }
    float bias2 = b2[n];
    float ls = 0.f, lq = 0.f;
#pragma unroll
    for (int mt = 0; mt < MT; ++mt) {
#pragma unroll
        for (int r = 0; r < 8; ++r) {
            int m = mt * 16 + r + 8 * g;
            float v = c2[mt][r] + bias2;
            v = v > 0.f ? v : 0.f;                     // outer ReLU
            long row = row0 + m;
            if (row < nNodes) {
                y[row * DD + n] = v;
                ls += v;
                lq += v * v;
            }
        }
    }
    atomicAdd(&sSum[n], ls);                           // 2 contributors per column
    atomicAdd(&sSq[n], lq);
    __syncthreads();

    if (tid < DD) {                                    // deterministic per-block partials
        pSum[(long)blockIdx.x * DD + tid] = sSum[tid];
        pSq [(long)blockIdx.x * DD + tid] = sSq[tid];
    }
}

// ---------------- kernel 4: reduce partials -> BN scale/shift --------------
__global__ __launch_bounds__(128) void gin_bnstats_kernel(const float* __restrict__ pSum,
                                                          const float* __restrict__ pSq,
                                                          const float* __restrict__ gamma,
                                                          const float* __restrict__ beta,
                                                          float* __restrict__ scaleB,
                                                          float* __restrict__ shiftB,
                                                          int nTiles, int nNodes) {
    int c = threadIdx.x;
    float s = 0.f, q = 0.f;
    for (int t = 0; t < nTiles; ++t) {                 // fixed order -> deterministic
        s += pSum[(long)t * DD + c];
        q += pSq [(long)t * DD + c];
    }
    float inv_n = 1.0f / (float)nNodes;
    float mean  = s * inv_n;
    float var   = q * inv_n - mean * mean;
    float sc    = gamma[c] * rsqrtf(var + 1e-5f);
    scaleB[c] = sc;
    shiftB[c] = beta[c] - mean * sc;
}

// ---------------- kernel 5: in-place normalize of d_out --------------------
__global__ __launch_bounds__(256) void gin_bnapply_kernel(float* __restrict__ y,
                                                          const float* __restrict__ scaleB,
                                                          const float* __restrict__ shiftB,
                                                          long n4) {
    long i = (long)blockIdx.x * blockDim.x + threadIdx.x;
    if (i >= n4) return;
    float4 v = ((float4*)y)[i];
    int c = (int)((i * 4) & (DD - 1));
    v.x = v.x * scaleB[c + 0] + shiftB[c + 0];
    v.y = v.y * scaleB[c + 1] + shiftB[c + 1];
    v.z = v.z * scaleB[c + 2] + shiftB[c + 2];
    v.w = v.w * scaleB[c + 3] + shiftB[c + 3];
    ((float4*)y)[i] = v;
}

extern "C" void kernel_launch(void* const* d_in, const int* in_sizes, int n_in,
                              void* d_out, int out_size, void* d_ws, size_t ws_size,
                              hipStream_t stream) {
    const float* feature = (const float*)d_in[0];
    const int*   edge    = (const int*)d_in[1];
    const float* W1      = (const float*)d_in[2];
    const float* b1      = (const float*)d_in[3];
    const float* W2      = (const float*)d_in[4];
    const float* b2      = (const float*)d_in[5];
    const float* gamma   = (const float*)d_in[6];
    const float* beta    = (const float*)d_in[7];

    int nNodes = in_sizes[0] / DD;             // 50000
    int nE     = in_sizes[1] / 2;              // 600000
    const int* src = edge;
    const int* dst = edge + nE;

    float* out = (float*)d_out;

    // workspace layout
    char*  ws     = (char*)d_ws;
    float* agg    = (float*)ws;                                    // nNodes*128 f32
    size_t aggB   = (size_t)nNodes * DD * sizeof(float);
    int    nTiles = (nNodes + ROWS_PER_BLOCK - 1) / ROWS_PER_BLOCK;
    float* pSum   = (float*)(ws + aggB);                           // nTiles*128
    float* pSq    = pSum + (size_t)nTiles * DD;                    // nTiles*128
    float* scaleB = pSq + (size_t)nTiles * DD;                     // 128
    float* shiftB = scaleB + DD;                                   // 128
    _Float16* W1t = (_Float16*)(shiftB + DD);                      // 128*128 f16
    _Float16* W2t = W1t + DD * DD;                                 // 128*128 f16

    long n4 = (long)nNodes * (DD / 4);

    gin_init_kernel<<<(int)((n4 + 255) / 256), 256, 0, stream>>>(feature, agg, n4);
    gin_cvtw_kernel<<<(DD * DD + 255) / 256, 256, 0, stream>>>(W1, W2, W1t, W2t);
    gin_scatter_kernel<<<(nE + 7) / 8, 256, 0, stream>>>(feature, src, dst, agg, nE);
    gin_mlp_wmma_kernel<<<nTiles, 256, 0, stream>>>(agg, W1t, b1, W2t, b2,
                                                    out, pSum, pSq, nNodes);
    gin_bnstats_kernel<<<1, 128, 0, stream>>>(pSum, pSq, gamma, beta,
                                              scaleB, shiftB, nTiles, nNodes);
    gin_bnapply_kernel<<<(int)((n4 + 255) / 256), 256, 0, stream>>>(out, scaleB, shiftB, n4);
}
